// OnnxSafeMultiheadAttention_3384434230088
// MI455X (gfx1250) — matrix operand order repaired and checked by reading, not compiled, not run on hardware
//
#include <hip/hip_runtime.h>

#define EDIM 1024
#define NH   16
#define DH   64
#define BB   8
#define TT   1024

typedef __attribute__((ext_vector_type(16))) _Float16 v16h;
typedef __attribute__((ext_vector_type(8)))  _Float16 v8h;
typedef __attribute__((ext_vector_type(8)))  float    v8f;

static __device__ inline v16h mk16(v8h lo, v8h hi) {
  return __builtin_shufflevector(lo, hi, 0,1,2,3,4,5,6,7,8,9,10,11,12,13,14,15);
}
static __device__ inline v8h cvt8(float4 a, float4 b) {
  v8h r;
  r[0]=(_Float16)a.x; r[1]=(_Float16)a.y; r[2]=(_Float16)a.z; r[3]=(_Float16)a.w;
  r[4]=(_Float16)b.x; r[5]=(_Float16)b.y; r[6]=(_Float16)b.z; r[7]=(_Float16)b.w;
  return r;
}
static __device__ inline v8f wmma16(v16h a, v16h b, v8f c) {
  // D = A(16x32 f16) x B(32x16 f16) + C(16x16 f32)
  return __builtin_amdgcn_wmma_f32_16x16x32_f16(false, a, false, b, (short)0, c, false, false);
}

// CDNA5 async global->LDS DMA (ASYNCcnt-tracked); 16 bytes per lane.
// LDS destination address comes from truncating the generic pointer: for the
// LDS aperture, addr[31:0] is the wave-relative LDS byte address (ISA 10.2).
static __device__ inline void async_ld16(const void* lds_dst, const void* gsrc) {
  unsigned lds_off = (unsigned)(unsigned long long)lds_dst;
  asm volatile("global_load_async_to_lds_b128 %0, %1, off"
               :: "v"(lds_off), "v"(gsrc) : "memory");
}
static __device__ inline void async_wait0() {
  asm volatile("s_wait_asynccnt 0x0" ::: "memory");
}

// ---------------------------------------------------------------------------
// Generic 128x128x32-tiled GEMM with register pipelining:
//   Y[m,n] = sum_k A[m,k]*W[n,k] + bias[n]
// A is fp32 (A_F16=false) or f16 (A_F16=true); W/bias fp32.
// mode 0: fp32 out[m*1024+n]
// mode 1: f16 scatter to [B,H,T,Dh]  (Q,K)
// mode 2: f16 scatter to [B,H,Dh,T]  (V, transposed for PV B-operand)
// (Register pipeline kept here: tiles need in-flight f32->f16 conversion,
//  which the async byte-copy path cannot do.)
// ---------------------------------------------------------------------------
template<bool A_F16>
__global__ __launch_bounds__(256) void gemm_wmma_kernel(
    const void* __restrict__ Ap, const float* __restrict__ W,
    const float* __restrict__ bias, void* __restrict__ Out, int mode)
{
  constexpr int KP = 48;  // LDS row pitch in halves: 96 B, 16B-aligned
  __shared__ _Float16 Xs[128][KP];
  __shared__ _Float16 Ws[128][KP];

  const int tid  = threadIdx.x;
  const int wave = tid >> 5, lane = tid & 31;
  const int lm = lane & 15, lh = lane >> 4;
  const int m0 = blockIdx.x * 128, n0 = blockIdx.y * 128;
  const int rowBase = (wave >> 2) * 64;   // wave grid 2x4 -> 64x32 per wave
  const int colBase = (wave & 3) * 32;
  const int loadRow = tid >> 1;           // 0..127
  const int loadCol = (tid & 1) * 16;     // 0 or 16

  const float*    Wrow = W + (size_t)(n0 + loadRow) * EDIM + loadCol;
  const float*    Af   = (const float*)Ap    + (size_t)(m0 + loadRow) * EDIM + loadCol;
  const _Float16* Ah   = (const _Float16*)Ap + (size_t)(m0 + loadRow) * EDIM + loadCol;

  v8f acc[4][2] = {};
  v8h ra0, ra1, rw0, rw1;

  auto stage_load = [&](int k0) {
    if constexpr (A_F16) {
      ra0 = *(const v8h*)(Ah + k0);
      ra1 = *(const v8h*)(Ah + k0 + 8);
    } else {
      const float4* s = (const float4*)(Af + k0);
      ra0 = cvt8(s[0], s[1]);
      ra1 = cvt8(s[2], s[3]);
    }
    const float4* w = (const float4*)(Wrow + k0);
    rw0 = cvt8(w[0], w[1]);
    rw1 = cvt8(w[2], w[3]);
  };

  stage_load(0);

  for (int k0 = 0; k0 < EDIM; k0 += 32) {
    *(v8h*)&Xs[loadRow][loadCol]     = ra0;
    *(v8h*)&Xs[loadRow][loadCol + 8] = ra1;
    *(v8h*)&Ws[loadRow][loadCol]     = rw0;
    *(v8h*)&Ws[loadRow][loadCol + 8] = rw1;
    __syncthreads();

    // Overlap next stage's global traffic with this stage's WMMAs.
    if (k0 + 32 < EDIM) {
      stage_load(k0 + 32);
      if (k0 + 64 < EDIM) {
        if constexpr (A_F16) __builtin_prefetch(Ah + k0 + 64, 0, 1);
        else                 __builtin_prefetch(Af + k0 + 64, 0, 1);
        __builtin_prefetch(Wrow + k0 + 64, 0, 1);
      }
    }

    v16h afr[4];
#pragma unroll
    for (int i = 0; i < 4; ++i) {
      const _Float16* rp = &Xs[rowBase + i*16 + lm][0];
      afr[i] = mk16(*(const v8h*)(rp + lh*8), *(const v8h*)(rp + 16 + lh*8));
    }
    v16h bfr[2];
#pragma unroll
    for (int j = 0; j < 2; ++j) {
      const _Float16* cp = &Ws[colBase + j*16 + lm][0];
      bfr[j] = mk16(*(const v8h*)(cp + lh*8), *(const v8h*)(cp + 16 + lh*8));
    }
#pragma unroll
    for (int i = 0; i < 4; ++i)
#pragma unroll
      for (int j = 0; j < 2; ++j)
        acc[i][j] = wmma16(afr[i], bfr[j], acc[i][j]);
    __syncthreads();
  }

#pragma unroll
  for (int i = 0; i < 4; ++i) {
#pragma unroll
    for (int j = 0; j < 2; ++j) {
      const int gcol = n0 + colBase + j*16 + lm;
      const float bv = bias[gcol];
#pragma unroll
      for (int r = 0; r < 8; ++r) {
        const int grow = m0 + rowBase + i*16 + lh*8 + r;
        const float v = acc[i][j][r] + bv;
        if (mode == 0) {
          ((float*)Out)[(size_t)grow * EDIM + gcol] = v;
        } else {
          const int bb = grow >> 10, t = grow & (TT - 1);
          const int hh = gcol >> 6,  d = gcol & (DH - 1);
          if (mode == 1)
            ((_Float16*)Out)[(((size_t)bb * NH + hh) * TT + t) * DH + d] = (_Float16)v;
          else
            ((_Float16*)Out)[(((size_t)bb * NH + hh) * DH + d) * TT + t] = (_Float16)v;
        }
      }
    }
  }
}

// ---------------------------------------------------------------------------
// Flash attention: one block = (b,h) x 64 query rows; stream 64-key tiles.
// Q/K/V/mask tiles are filled with async global->LDS DMA (ASYNCcnt),
// letting the tile fills proceed while WMMAs of the previous phase issue.
// ---------------------------------------------------------------------------
__global__ __launch_bounds__(256) void attn_kernel(
    const _Float16* __restrict__ Qw, const _Float16* __restrict__ Kw,
    const _Float16* __restrict__ Vw, const float* __restrict__ amask,
    const unsigned char* __restrict__ kpm, _Float16* __restrict__ Ow)
{
  constexpr int P = 72;  // 144 B pitch, 16B-aligned
  __shared__ _Float16 Qs[64][P];
  __shared__ _Float16 Ks[64][P];
  __shared__ _Float16 Vs[64][P];   // Vs[d][k] (transposed in workspace)
  __shared__ _Float16 Ps[64][P];
  __shared__ float Ss[64][68];     // 272 B pitch, 16B-aligned
  __shared__ float mrow[64], lrow[64], arow[64];

  const int tid  = threadIdx.x;
  const int wave = tid >> 5, lane = tid & 31;
  const int lm = lane & 15, lh = lane >> 4;
  const int bh = blockIdx.y;
  const int b = bh >> 4, hd = bh & 15;
  const int qb = blockIdx.x * 64;
  const size_t base  = (size_t)bh * TT * DH;
  const size_t vbase = (size_t)bh * DH * TT;
  const int qs = (wave & 3) * 16;   // wave's 16-row q block
  const int ch = (wave >> 2) * 32;  // wave's 32-col block (keys for S, d for O)
  const int lr = tid >> 2, lc = (tid & 3) * 16;

  {
    const _Float16* src = Qw + base + (size_t)(qb + lr) * DH + lc;
    async_ld16(&Qs[lr][lc],     src);
    async_ld16(&Qs[lr][lc + 8], src + 8);
  }
  if (tid < 64) { mrow[tid] = -1e30f; lrow[tid] = 0.0f; }
  v8f oacc[2] = {};
  async_wait0();
  __syncthreads();

  const float scale = 0.125f;  // Dh^-0.5

  for (int kt = 0; kt < TT; kt += 64) {
    {
      const _Float16* ksrc = Kw + base + (size_t)(kt + lr) * DH + lc;
      async_ld16(&Ks[lr][lc],     ksrc);
      async_ld16(&Ks[lr][lc + 8], ksrc + 8);
      const _Float16* vsrc = Vw + vbase + (size_t)lr * TT + kt + lc;
      async_ld16(&Vs[lr][lc],     vsrc);
      async_ld16(&Vs[lr][lc + 8], vsrc + 8);
      // mask tile -> Ss verbatim (64 contiguous bytes per thread)
      const float* msrc = amask + (size_t)(qb + lr) * TT + kt + lc;
      async_ld16(&Ss[lr][lc],      msrc);
      async_ld16(&Ss[lr][lc + 4],  msrc + 4);
      async_ld16(&Ss[lr][lc + 8],  msrc + 8);
      async_ld16(&Ss[lr][lc + 12], msrc + 12);
    }
    // prefetch next tiles into L2/WGP$ while this tile computes
    if (kt + 64 < TT) {
      __builtin_prefetch(Kw + base + (size_t)(kt + 64 + lr) * DH + lc, 0, 1);
      __builtin_prefetch(Vw + vbase + (size_t)lr * TT + kt + 64 + lc, 0, 1);
      __builtin_prefetch(amask + (size_t)(qb + lr) * TT + kt + 64 + lc, 0, 1);
    }
    async_wait0();
    __syncthreads();

    // S = (Q K^T)*scale + attn_mask (+ padding mask), accumulated into Ss
#pragma unroll
    for (int j = 0; j < 2; ++j) {
      v8f s = {};
#pragma unroll
      for (int kk = 0; kk < 2; ++kk) {
        const _Float16* qp = &Qs[qs + lm][kk*32];
        v16h af = mk16(*(const v8h*)(qp + lh*8), *(const v8h*)(qp + 16 + lh*8));
        const _Float16* kp = &Ks[ch + j*16 + lm][kk*32];
        v16h bf = mk16(*(const v8h*)(kp + lh*8), *(const v8h*)(kp + 16 + lh*8));
        s = wmma16(af, bf, s);
      }
      const int coll = ch + j*16 + lm;
      const int gk = kt + coll;
      const float padv = kpm[b * TT + gk] ? -1e30f : 0.0f;
#pragma unroll
      for (int r = 0; r < 8; ++r) {
        const int rowl = qs + lh*8 + r;
        Ss[rowl][coll] = s[r] * scale + Ss[rowl][coll] + padv;  // same-thread RMW
      }
    }
    __syncthreads();

    // online softmax bookkeeping, one row per thread (tid < 64)
    if (tid < 64) {
      const float mo = mrow[tid];
      float mx = mo;
#pragma unroll 8
      for (int c = 0; c < 64; ++c) mx = fmaxf(mx, Ss[tid][c]);
      const float al = __expf(mo - mx);
      float sum = 0.0f;
#pragma unroll 8
      for (int c = 0; c < 64; ++c) {
        const float p = __expf(Ss[tid][c] - mx);
        Ps[tid][c] = (_Float16)p;
        sum += p;
      }
      mrow[tid] = mx;
      lrow[tid] = lrow[tid] * al + sum;
      arow[tid] = al;
    }
    __syncthreads();

    // O = O*alpha + P V
#pragma unroll
    for (int j = 0; j < 2; ++j) {
#pragma unroll
      for (int r = 0; r < 8; ++r)
        oacc[j][r] *= arow[qs + lh*8 + r];
#pragma unroll
      for (int kk = 0; kk < 2; ++kk) {
        const _Float16* pp = &Ps[qs + lm][kk*32];
        v16h af = mk16(*(const v8h*)(pp + lh*8), *(const v8h*)(pp + 16 + lh*8));
        const _Float16* vp = &Vs[ch + j*16 + lm][kk*32];
        v16h bf = mk16(*(const v8h*)(vp + lh*8), *(const v8h*)(vp + 16 + lh*8));
        oacc[j] = wmma16(af, bf, oacc[j]);
      }
    }
    __syncthreads();
  }

#pragma unroll
  for (int j = 0; j < 2; ++j) {
    const int d = ch + j*16 + lm;
#pragma unroll
    for (int r = 0; r < 8; ++r) {
      const int rowl = qs + lh*8 + r;
      const float ov = oacc[j][r] / lrow[rowl];
      Ow[(size_t)(b * TT + qb + rowl) * EDIM + hd * DH + d] = (_Float16)ov;
    }
  }
}

extern "C" void kernel_launch(void* const* d_in, const int* in_sizes, int n_in,
                              void* d_out, int out_size, void* d_ws, size_t ws_size,
                              hipStream_t stream) {
  (void)in_sizes; (void)n_in; (void)out_size; (void)ws_size;
  const float* query = (const float*)d_in[0];
  const float* key   = (const float*)d_in[1];
  const float* value = (const float*)d_in[2];
  const unsigned char* kpm = (const unsigned char*)d_in[3];
  const float* amask = (const float*)d_in[4];
  const float* w_in  = (const float*)d_in[5];
  const float* b_in  = (const float*)d_in[6];
  const float* w_out = (const float*)d_in[7];
  const float* b_out = (const float*)d_in[8];

  char* ws = (char*)d_ws;
  const size_t SEG = (size_t)1 << 24;  // 16 MB: B*H*T*Dh f16
  _Float16* q16 = (_Float16*)(ws);
  _Float16* k16 = (_Float16*)(ws + SEG);
  _Float16* v16 = (_Float16*)(ws + 2 * SEG);
  _Float16* o16 = (_Float16*)(ws + 3 * SEG);

  dim3 blk(256);
  dim3 gg(BB * TT / 128, EDIM / 128);  // (64, 8)

  gemm_wmma_kernel<false><<<gg, blk, 0, stream>>>(query, w_in,               b_in,            q16, 1);
  gemm_wmma_kernel<false><<<gg, blk, 0, stream>>>(key,   w_in + EDIM*EDIM,   b_in + EDIM,     k16, 1);
  gemm_wmma_kernel<false><<<gg, blk, 0, stream>>>(value, w_in + 2*EDIM*EDIM, b_in + 2*EDIM,   v16, 2);

  attn_kernel<<<dim3(TT / 64, BB * NH), blk, 0, stream>>>(q16, k16, v16, amask, kpm, o16);

  gemm_wmma_kernel<true><<<gg, blk, 0, stream>>>(o16, w_out, b_out, d_out, 0);
}
